// SecureSwishFeedForward_70214125355102
// MI455X (gfx1250) — matrix operand order repaired and checked by reading, not compiled, not run on hardware
//
#include <hip/hip_runtime.h>
#include <stddef.h>
#include <stdint.h>

// ---------------------------------------------------------------------------
// SecureSwishFeedForward for MI455X (gfx1250, wave32, WMMA).
//   out = (swishpoly(x @ W1^T + b1)) @ W2^T + b2
// Fast path: one f32->bf16 conversion pass for x/W1/W2 into d_ws, then two
// bf16 GEMM passes using global_load_async_to_lds_b128 double buffering and
// v_wmma_f32_16x16x32_bf16 (f32 accumulation). Intermediate activation is
// stored bf16 in d_ws (chunked over M to fit any ws_size).
// Fallback path (small ws): round-1 kernel converting f32->bf16 in-kernel.
// ---------------------------------------------------------------------------

typedef __attribute__((ext_vector_type(16))) __bf16 v16bf_t;
typedef __attribute__((ext_vector_type(8)))  __bf16 bf16x8_t;
typedef __attribute__((ext_vector_type(8)))  float  v8f_t;
typedef __attribute__((ext_vector_type(4)))  int    v4i_t;

union V16 {
  v16bf_t  v;
  bf16x8_t h[2];
};

#define BM 128
#define BN 128

// fallback (convert-in-kernel) path
#define BK1  32
#define LDT1 40   // 32+8 pad: 80B row stride (16B aligned, conflict-free)

// async bf16 path
#define BK2  64
#define LDT2 72   // 64+8 pad: 144B row stride (16B aligned, conflict-free)

// round-to-nearest-even f32 -> bf16 (bit pattern)
__device__ __forceinline__ unsigned short f2bf(float f) {
  unsigned int u = __builtin_bit_cast(unsigned int, f);
  u += 0x7FFFu + ((u >> 16) & 1u);
  return (unsigned short)(u >> 16);
}

// ---------------------------------------------------------------------------
// Async global->LDS copy via direct LLVM intrinsic declaration (CK-style;
// sidesteps the builtin's address-space qualifier spelling). Signature at the
// LLVM level: (ptr addrspace(1), ptr addrspace(3), i32 imm offset, i32 cpol).
// ---------------------------------------------------------------------------
__device__ void
llvm_amdgcn_global_load_async_to_lds_b128(
    const __attribute__((address_space(1))) v4i_t* gsrc,
    __attribute__((address_space(3))) v4i_t* ldst,
    int offset, int cpol) __asm("llvm.amdgcn.global.load.async.to.lds.b128");

// 16B global -> LDS async copy (ASYNCcnt-tracked, no VGPR data).
// Generic LDS address: low 32 bits == LDS offset (aperture layout).
__device__ __forceinline__ void async_cp16(const unsigned short* g, unsigned short* l) {
  const __attribute__((address_space(1))) v4i_t* gp =
      (const __attribute__((address_space(1))) v4i_t*)(uintptr_t)g;
  __attribute__((address_space(3))) v4i_t* lp =
      (__attribute__((address_space(3))) v4i_t*)(unsigned int)(uintptr_t)l;
  llvm_amdgcn_global_load_async_to_lds_b128(gp, lp, 0, 0);
}

template <int N>
__device__ __forceinline__ void wait_async() {
#if __has_builtin(__builtin_amdgcn_s_wait_asynccnt)
  __builtin_amdgcn_s_wait_asynccnt(N);
#else
  asm volatile("s_wait_asynccnt %0" :: "n"(N) : "memory");
#endif
}

// ---------------------------------------------------------------------------
// Pass 0: f32 -> bf16 conversion (bandwidth bound).
// ---------------------------------------------------------------------------
__global__ __launch_bounds__(256)
void cvt_f32_to_bf16(const float* __restrict__ s, unsigned short* __restrict__ d, int n4) {
  int i = blockIdx.x * blockDim.x + threadIdx.x;
  if (i >= n4) return;
  float4 v = ((const float4*)s)[i];
  unsigned int lo = (unsigned int)f2bf(v.x) | ((unsigned int)f2bf(v.y) << 16);
  unsigned int hi = (unsigned int)f2bf(v.z) | ((unsigned int)f2bf(v.w) << 16);
  uint2 o; o.x = lo; o.y = hi;
  *(uint2*)(d + (size_t)i * 4) = o;
}

// ---------------------------------------------------------------------------
// Fast path GEMM: all-bf16 operands, async-to-LDS double buffering, BK=64.
//   C[M,N] = act(A[M,K] @ W[N,K]^T + bias)
// SWISH: C is bf16 with fused bias + polynomial swish; else C is f32 + bias.
// ---------------------------------------------------------------------------
template <bool SWISH>
__global__ __launch_bounds__(256)
void ffn_gemm_bf16_async(const unsigned short* __restrict__ A,  // [M,K] bf16
                         const unsigned short* __restrict__ W,  // [N,K] bf16
                         const float* __restrict__ bias,        // [N]
                         void* __restrict__ Cptr,
                         int Mdim, int Kdim, int Ndim) {
  __shared__ unsigned short as[2][BM * LDT2];
  __shared__ unsigned short bs[2][BN * LDT2];

  const int tid   = threadIdx.x;
  const int lane  = tid & 31;
  const int wid   = tid >> 5;
  const int lmod  = lane & 15;
  const int lhalf = lane >> 4;

  const int m0 = blockIdx.y * BM;
  const int n0 = blockIdx.x * BN;
  const int wm = (wid & 3) * 32;   // 4 waves along M
  const int wn = (wid >> 2) * 64;  // 2 waves along N

  const v8f_t vzero = {0.f, 0.f, 0.f, 0.f, 0.f, 0.f, 0.f, 0.f};
  v8f_t acc[2][4];
#pragma unroll
  for (int i = 0; i < 2; ++i)
#pragma unroll
    for (int j = 0; j < 4; ++j) acc[i][j] = vzero;

  // Stage one 128x64 bf16 tile pair: 8 async b128 copies per thread.
  auto stage = [&](int buf, int k0) {
#pragma unroll
    for (int t = 0; t < 4; ++t) {
      const int idx = tid + t * 256;      // 1024 chunks of 8 bf16
      const int r   = idx >> 3;
      const int c   = (idx & 7) * 8;
      async_cp16(A + (size_t)(m0 + r) * Kdim + k0 + c, &as[buf][r * LDT2 + c]);
    }
#pragma unroll
    for (int t = 0; t < 4; ++t) {
      const int idx = tid + t * 256;
      const int r   = idx >> 3;
      const int c   = (idx & 7) * 8;
      async_cp16(W + (size_t)(n0 + r) * Kdim + k0 + c, &bs[buf][r * LDT2 + c]);
    }
  };

  stage(0, 0);
  const int nk = Kdim / BK2;
  for (int it = 0; it < nk; ++it) {
    const int  cur = it & 1;
    const bool nxt = (it + 1) < nk;
    if (nxt) stage(cur ^ 1, (it + 1) * BK2);
    // ASYNCcnt is in-order: <=8 outstanding means current buffer's 8 landed.
    if (nxt) wait_async<8>(); else wait_async<0>();
    __syncthreads();

#pragma unroll
    for (int ks = 0; ks < 2; ++ks) {
      const int kc = ks * 32;
      // A fragment (ISA 7.12.2): lane<16 -> K[0..8)+[16..24); lane>=16 -> +8.
      V16 a[2];
#pragma unroll
      for (int i = 0; i < 2; ++i) {
        const unsigned short* ap =
            &as[cur][(wm + i * 16 + lmod) * LDT2 + kc + lhalf * 8];
        a[i].h[0] = *(const bf16x8_t*)ap;
        a[i].h[1] = *(const bf16x8_t*)(ap + 16);
      }
      // B fragment: lane%16 = N column; lane<16 -> K[0..8)+[16..24); lane>=16 -> +8.
      V16 b[4];
#pragma unroll
      for (int j = 0; j < 4; ++j) {
        const unsigned short* bp =
            &bs[cur][(wn + j * 16 + lmod) * LDT2 + kc + lhalf * 8];
        b[j].h[0] = *(const bf16x8_t*)bp;
        b[j].h[1] = *(const bf16x8_t*)(bp + 16);
      }
#pragma unroll
      for (int i = 0; i < 2; ++i)
#pragma unroll
        for (int j = 0; j < 4; ++j)
          acc[i][j] = __builtin_amdgcn_wmma_f32_16x16x32_bf16(
              false, a[i].v, false, b[j].v, (short)0, acc[i][j], false, false);
    }
    __syncthreads();  // readers done before this buffer is restaged
  }

  // Epilogue. C layout: VGPR r -> M = r + 8*lhalf, N = lmod.
#pragma unroll
  for (int i = 0; i < 2; ++i) {
#pragma unroll
    for (int j = 0; j < 4; ++j) {
      const int n  = n0 + wn + j * 16 + lmod;
      const float bn = bias[n];
      const int mb = m0 + wm + i * 16 + lhalf * 8;
      if (SWISH) {
        unsigned short* C = (unsigned short*)Cptr;
#pragma unroll
        for (int r = 0; r < 8; ++r) {
          float h  = acc[i][j][r] + bn;
          float sw = h * (0.5f + 0.25f * h - 0.0208f * (h * h * h));
          C[(size_t)(mb + r) * Ndim + n] = f2bf(sw);
        }
      } else {
        float* C = (float*)Cptr;
#pragma unroll
        for (int r = 0; r < 8; ++r)
          C[(size_t)(mb + r) * Ndim + n] = acc[i][j][r] + bn;
      }
    }
  }
}

// ---------------------------------------------------------------------------
// Fallback path (round-1): convert f32->bf16 in-kernel, BK=32, no async.
// ---------------------------------------------------------------------------
__device__ __forceinline__ void stage_f32_tile(const float* __restrict__ src,
                                               size_t ld, unsigned short* dst,
                                               int tid) {
  const int e = tid * 16;
  const int r = e >> 5;
  const int c = e & 31;
  const float* p = src + (size_t)r * ld + c;
  unsigned short* q = dst + r * LDT1 + c;
#pragma unroll
  for (int i = 0; i < 4; ++i) {
    float4 v = *(const float4*)(p + i * 4);
    q[i * 4 + 0] = f2bf(v.x);
    q[i * 4 + 1] = f2bf(v.y);
    q[i * 4 + 2] = f2bf(v.z);
    q[i * 4 + 3] = f2bf(v.w);
  }
}

__device__ __forceinline__ void stage_bf16_tile(const unsigned short* __restrict__ src,
                                                size_t ld, unsigned short* dst,
                                                int tid) {
  const int e = tid * 16;
  const int r = e >> 5;
  const int c = e & 31;
  const uint4* p = (const uint4*)(src + (size_t)r * ld + c);
  uint4 v0 = p[0];
  uint4 v1 = p[1];
  *(uint4*)(dst + r * LDT1 + c)     = v0;
  *(uint4*)(dst + r * LDT1 + c + 8) = v1;
}

template <bool STAGE1>
__global__ __launch_bounds__(256)
void ffn_wmma_gemm(const void*  __restrict__ Aptr,
                   const float* __restrict__ Wptr,
                   const float* __restrict__ bias,
                   void* __restrict__ Cptr,
                   int Mdim, int Kdim, int Ndim) {
  __shared__ unsigned short as[BM * LDT1];
  __shared__ unsigned short bs[BN * LDT1];

  const int tid   = threadIdx.x;
  const int lane  = tid & 31;
  const int wid   = tid >> 5;
  const int lmod  = lane & 15;
  const int lhalf = lane >> 4;

  const int m0 = blockIdx.y * BM;
  const int n0 = blockIdx.x * BN;
  const int wm = (wid & 3) * 32;
  const int wn = (wid >> 2) * 64;

  const v8f_t vzero = {0.f, 0.f, 0.f, 0.f, 0.f, 0.f, 0.f, 0.f};
  v8f_t acc[2][4];
#pragma unroll
  for (int i = 0; i < 2; ++i)
#pragma unroll
    for (int j = 0; j < 4; ++j) acc[i][j] = vzero;

  for (int k0 = 0; k0 < Kdim; k0 += BK1) {
    __syncthreads();
    if (STAGE1)
      stage_f32_tile((const float*)Aptr + (size_t)m0 * Kdim + k0, (size_t)Kdim, as, tid);
    else
      stage_bf16_tile((const unsigned short*)Aptr + (size_t)m0 * Kdim + k0, (size_t)Kdim, as, tid);
    stage_f32_tile(Wptr + (size_t)n0 * Kdim + k0, (size_t)Kdim, bs, tid);
    __syncthreads();

    V16 a[2];
#pragma unroll
    for (int i = 0; i < 2; ++i) {
      const unsigned short* ap = &as[(wm + i * 16 + lmod) * LDT1 + lhalf * 8];
      a[i].h[0] = *(const bf16x8_t*)ap;
      a[i].h[1] = *(const bf16x8_t*)(ap + 16);
    }
    V16 b[4];
#pragma unroll
    for (int j = 0; j < 4; ++j) {
      const unsigned short* bp = &bs[(wn + j * 16 + lmod) * LDT1 + lhalf * 8];
      b[j].h[0] = *(const bf16x8_t*)bp;
      b[j].h[1] = *(const bf16x8_t*)(bp + 16);
    }
#pragma unroll
    for (int i = 0; i < 2; ++i)
#pragma unroll
      for (int j = 0; j < 4; ++j)
        acc[i][j] = __builtin_amdgcn_wmma_f32_16x16x32_bf16(
            false, a[i].v, false, b[j].v, (short)0, acc[i][j], false, false);
  }

#pragma unroll
  for (int i = 0; i < 2; ++i) {
#pragma unroll
    for (int j = 0; j < 4; ++j) {
      const int n  = n0 + wn + j * 16 + lmod;
      const float bn = bias[n];
      const int mb = m0 + wm + i * 16 + lhalf * 8;
      if (STAGE1) {
        unsigned short* C = (unsigned short*)Cptr;
#pragma unroll
        for (int r = 0; r < 8; ++r) {
          float h  = acc[i][j][r] + bn;
          float sw = h * (0.5f + 0.25f * h - 0.0208f * (h * h * h));
          C[(size_t)(mb + r) * Ndim + n] = f2bf(sw);
        }
      } else {
        float* C = (float*)Cptr;
#pragma unroll
        for (int r = 0; r < 8; ++r)
          C[(size_t)(mb + r) * Ndim + n] = acc[i][j][r] + bn;
      }
    }
  }
}

// ---------------------------------------------------------------------------
// Host launcher.
// ---------------------------------------------------------------------------
extern "C" void kernel_launch(void* const* d_in, const int* in_sizes, int n_in,
                              void* d_out, int out_size, void* d_ws, size_t ws_size,
                              hipStream_t stream) {
  const float* X  = (const float*)d_in[0];  // [B,S,D] f32
  const float* W1 = (const float*)d_in[1];  // [F,D]   f32
  const float* b1 = (const float*)d_in[2];  // [F]
  const float* W2 = (const float*)d_in[3];  // [D,F]   f32
  const float* b2 = (const float*)d_in[4];  // [D]

  const int F = in_sizes[2];      // 4096
  const int D = in_sizes[4];      // 1024
  const int M = in_sizes[0] / D;  // B*S = 8192

  const size_t eW1 = (size_t)F * D;
  const size_t eW2 = (size_t)D * F;
  const size_t eX  = (size_t)M * D;
  const size_t base_bytes = (eW1 + eW2 + eX) * 2;
  const size_t row_bytes  = (size_t)F * 2;

  dim3 blk(256, 1, 1);

  if (ws_size >= base_bytes + row_bytes * (size_t)BM) {
    // ---- fast path: pre-convert operands to bf16, async WMMA GEMMs ----
    unsigned short* W1b = (unsigned short*)d_ws;
    unsigned short* W2b = W1b + eW1;
    unsigned short* Xb  = W2b + eW2;
    unsigned short* C1  = Xb + eX;

    const size_t rem = ws_size - base_bytes;
    int mc = (int)(((rem / row_bytes) / BM) * BM);
    if (mc < BM) mc = BM;
    if (mc > M)  mc = M;

    {
      int n4 = (int)(eW1 / 4);
      hipLaunchKernelGGL(cvt_f32_to_bf16, dim3((n4 + 255) / 256), blk, 0, stream, W1, W1b, n4);
      n4 = (int)(eW2 / 4);
      hipLaunchKernelGGL(cvt_f32_to_bf16, dim3((n4 + 255) / 256), blk, 0, stream, W2, W2b, n4);
      n4 = (int)(eX / 4);
      hipLaunchKernelGGL(cvt_f32_to_bf16, dim3((n4 + 255) / 256), blk, 0, stream, X, Xb, n4);
    }

    for (int ms = 0; ms < M; ms += mc) {
      const int mcur = (M - ms < mc) ? (M - ms) : mc;
      dim3 g1(F / BN, mcur / BM, 1);
      hipLaunchKernelGGL((ffn_gemm_bf16_async<true>), g1, blk, 0, stream,
                         Xb + (size_t)ms * D, W1b, b1, (void*)C1, mcur, D, F);
      dim3 g2(D / BN, mcur / BM, 1);
      hipLaunchKernelGGL((ffn_gemm_bf16_async<false>), g2, blk, 0, stream,
                         C1, W2b, b2, (void*)((float*)d_out + (size_t)ms * D),
                         mcur, F, D);
    }
  } else {
    // ---- fallback: convert-in-kernel path, chunk intermediate over M ----
    unsigned short* C1 = (unsigned short*)d_ws;
    long long rows = (ws_size > 0) ? (long long)(ws_size / row_bytes) : 0;
    int mc = (int)((rows / BM) * BM);
    if (mc < BM) mc = BM;
    if (mc > M)  mc = M;

    for (int ms = 0; ms < M; ms += mc) {
      const int mcur = (M - ms < mc) ? (M - ms) : mc;
      dim3 g1(F / BN, mcur / BM, 1);
      hipLaunchKernelGGL((ffn_wmma_gemm<true>), g1, blk, 0, stream,
                         (const void*)(X + (size_t)ms * D), W1, b1,
                         (void*)C1, mcur, D, F);
      dim3 g2(D / BN, mcur / BM, 1);
      hipLaunchKernelGGL((ffn_wmma_gemm<false>), g2, blk, 0, stream,
                         (const void*)C1, W2, b2,
                         (void*)((float*)d_out + (size_t)ms * D), mcur, F, D);
    }
  }
}
